// MixtureOfExperts_55448027791425
// MI455X (gfx1250) — matrix operand order repaired and checked by reading, not compiled, not run on hardware
//
#include <hip/hip_runtime.h>
#include <hip/hip_bf16.h>

#define N_TOK 8192
#define DIM   1024
#define NEXP  10

#define TM 128
#define TN 128
#define TK 32
#define LDK 40   // padded K stride (elements) -> 80B rows, 16B aligned

typedef __attribute__((ext_vector_type(16))) __bf16 v16bf;
typedef __attribute__((ext_vector_type(8)))  __bf16 v8bf;
typedef __attribute__((ext_vector_type(8)))  float  v8f;
typedef __attribute__((__vector_size__(4 * sizeof(int)))) int v4i;

union Frag16 { v16bf v; v8bf h[2]; };

#if __has_builtin(__builtin_amdgcn_global_load_async_to_lds_b128)
#define MOE_HAS_ASYNC 1
#else
#define MOE_HAS_ASYNC 0
#endif

__device__ __forceinline__ void moe_wait_async() {
#if MOE_HAS_ASYNC
#if __has_builtin(__builtin_amdgcn_s_wait_asynccnt)
    __builtin_amdgcn_s_wait_asynccnt(0);
#else
    asm volatile("s_wait_asynccnt 0x0" ::: "memory");
#endif
#endif
}

// ---------------- small init ----------------
__global__ void moe_init_counts(int* counts) {
    if (threadIdx.x < NEXP) counts[threadIdx.x] = 0;
}

// ------------- We (f32, [E][Din][Dout]) -> WeT (bf16, [E][Dout][Din]) -------------
__global__ void moe_we_transpose(const float* __restrict__ We, __bf16* __restrict__ WeT) {
    __shared__ float tile[64][65];
    const int b   = blockIdx.x;
    const int e   = b >> 8;
    const int rem = b & 255;
    const int d0  = (rem >> 4) << 6;
    const int f0  = (rem & 15) << 6;
    const float* src = We  + ((size_t)e << 20);
    __bf16*      dst = WeT + ((size_t)e << 20);
    #pragma unroll
    for (int i = 0; i < 16; ++i) {
        int idx = threadIdx.x + 256 * i;
        int r = idx >> 6, c = idx & 63;
        tile[r][c] = src[(size_t)(d0 + r) * DIM + f0 + c];
    }
    __syncthreads();
    #pragma unroll
    for (int i = 0; i < 16; ++i) {
        int idx = threadIdx.x + 256 * i;
        int r = idx >> 6, c = idx & 63;
        dst[(size_t)(f0 + r) * DIM + d0 + c] = (__bf16)tile[c][r];
    }
}

// ------------- router: top-2 experts per token + x -> bf16 -------------
__global__ void moe_router(const float* __restrict__ x,
                           const float* __restrict__ Wr,
                           const float* __restrict__ br,
                           __bf16* __restrict__ xbf,
                           int* __restrict__ topk_idx,
                           int* __restrict__ counts) {
    const int wave = threadIdx.x >> 5;
    const int lane = threadIdx.x & 31;
    const int n    = blockIdx.x * 8 + wave;

    float2 xr[16];
    const float2* xp = (const float2*)(x + (size_t)n * DIM);
    #pragma unroll
    for (int j = 0; j < 16; ++j) xr[j] = xp[lane + 32 * j];

    unsigned int* xb = (unsigned int*)(xbf + (size_t)n * DIM);
    #pragma unroll
    for (int j = 0; j < 16; ++j) {
        __bf16 lo = (__bf16)xr[j].x, hi = (__bf16)xr[j].y;
        unsigned short ulo = *(unsigned short*)&lo, uhi = *(unsigned short*)&hi;
        xb[lane + 32 * j] = ((unsigned int)uhi << 16) | (unsigned int)ulo;
    }

    float best = -3.402823466e38f, second = -3.402823466e38f;
    int   bi = 0, si = 0;
    for (int e = 0; e < NEXP; ++e) {
        const float2* wp = (const float2*)(Wr + (size_t)e * DIM);
        float s = 0.f;
        #pragma unroll
        for (int j = 0; j < 16; ++j) {
            float2 w = wp[lane + 32 * j];
            s += xr[j].x * w.x + xr[j].y * w.y;
        }
        #pragma unroll
        for (int off = 16; off > 0; off >>= 1) s += __shfl_xor(s, off, 32);
        s += br[e];
        if (s > best)        { second = best; si = bi; best = s; bi = e; }
        else if (s > second) { second = s; si = e; }
    }
    if (lane == 0) {
        topk_idx[2 * n]     = bi;
        topk_idx[2 * n + 1] = si;
        atomicAdd(&counts[bi], 1);
        atomicAdd(&counts[si], 1);
    }
}

// ------------- tiny exclusive scan over 10 experts -------------
__global__ void moe_scan(const int* __restrict__ counts, int* offsets, int* cursors) {
    if (threadIdx.x == 0) {
        int acc = 0;
        for (int e = 0; e < NEXP; ++e) { offsets[e] = acc; cursors[e] = acc; acc += counts[e]; }
        offsets[NEXP] = acc;
    }
}

// ------------- scatter token ids into per-expert contiguous lists -------------
__global__ void moe_scatter(const int* __restrict__ topk_idx,
                            int* __restrict__ cursors,
                            int* __restrict__ token_list) {
    int n = blockIdx.x * blockDim.x + threadIdx.x;
    if (n >= N_TOK) return;
    int e0 = topk_idx[2 * n], e1 = topk_idx[2 * n + 1];
    token_list[atomicAdd(&cursors[e0], 1)] = n;
    token_list[atomicAdd(&cursors[e1], 1)] = n;
}

// ------------- zero output -------------
__global__ void moe_zero_out(float4* __restrict__ out) {
    out[(size_t)blockIdx.x * 256 + threadIdx.x] = make_float4(0.f, 0.f, 0.f, 0.f);
}

// ------------- grouped GEMM: 128x128 tile, 8 waves x (32x64), K-step 32 -------------
// Double-buffered LDS, one barrier per K-step, async global->LDS staging when available.
__global__ void moe_gemm(const __bf16* __restrict__ xbf,
                         const __bf16* __restrict__ WeT,
                         const float*  __restrict__ be,
                         const int*    __restrict__ counts,
                         const int*    __restrict__ offsets,
                         const int*    __restrict__ token_list,
                         float*        __restrict__ out) {
    __shared__ int toks[TM];
    __shared__ __align__(16) __bf16 As[2][TM][LDK];   // [buf][row][k]
    __shared__ __align__(16) __bf16 Bs[2][TN][LDK];   // [buf][col][k] (B^T)

    const int blk   = blockIdx.x;
    const int e     = blk >> 9;           // / (64*8)
    const int rem   = blk & 511;
    const int mtile = rem >> 3;
    const int ntile = rem & 7;
    const int cnt   = counts[e];
    const int m0    = mtile << 7;
    if (m0 >= cnt) return;
    const int base  = offsets[e];
    const int valid = min(TM, cnt - m0);
    const int n0    = ntile << 7;
    const int tid   = threadIdx.x;

    if (tid < TM) toks[tid] = token_list[base + m0 + min(tid, valid - 1)];
    __syncthreads();

    // staging role: each thread covers 32B of one A row and one B^T row
    const int srow = tid >> 1;            // 0..127
    const int skb  = (tid & 1) << 4;      // 0 or 16 (elements)
    const __bf16* agp = xbf + (size_t)toks[srow] * DIM + skb;
    const __bf16* bgp = WeT + ((size_t)e << 20) + (size_t)(n0 + srow) * DIM + skb;

    auto stage = [&](int k0, int buf) {
#if MOE_HAS_ASYNC
        __builtin_amdgcn_global_load_async_to_lds_b128(
            (v4i*)(agp + k0),     (v4i*)&As[buf][srow][skb],     0, 0);
        __builtin_amdgcn_global_load_async_to_lds_b128(
            (v4i*)(agp + k0 + 8), (v4i*)&As[buf][srow][skb + 8], 0, 0);
        __builtin_amdgcn_global_load_async_to_lds_b128(
            (v4i*)(bgp + k0),     (v4i*)&Bs[buf][srow][skb],     0, 0);
        __builtin_amdgcn_global_load_async_to_lds_b128(
            (v4i*)(bgp + k0 + 8), (v4i*)&Bs[buf][srow][skb + 8], 0, 0);
#else
        v8bf a0 = *(const v8bf*)(agp + k0);
        v8bf a1 = *(const v8bf*)(agp + k0 + 8);
        v8bf b0 = *(const v8bf*)(bgp + k0);
        v8bf b1 = *(const v8bf*)(bgp + k0 + 8);
        *(v8bf*)&As[buf][srow][skb]     = a0;
        *(v8bf*)&As[buf][srow][skb + 8] = a1;
        *(v8bf*)&Bs[buf][srow][skb]     = b0;
        *(v8bf*)&Bs[buf][srow][skb + 8] = b1;
#endif
    };

    const int wv   = tid >> 5;            // wave 0..7
    const int wm   = wv & 3;              // rows wm*32 .. +32
    const int wn   = wv >> 2;             // cols wn*64 .. +64
    const int lane = tid & 31;
    const int lh   = lane & 15;
    const int hi16 = lane >> 4;

    v8f acc[2][4] = {};

    stage(0, 0);
    moe_wait_async();
    __syncthreads();

    int buf = 0;
    for (int k0 = 0; k0 < DIM; k0 += TK) {
        if (k0 + TK < DIM) stage(k0 + TK, buf ^ 1);

        // A fragments (16x32 bf16): lane<16 -> K {0..7}u{16..23}; lane>=16 -> +8
        Frag16 af[2];
        #pragma unroll
        for (int s = 0; s < 2; ++s) {
            int row = (wm << 5) + (s << 4) + lh;
            int kb  = hi16 << 3;
            af[s].h[0] = *(const v8bf*)&As[buf][row][kb];
            af[s].h[1] = *(const v8bf*)&As[buf][row][kb + 16];
        }
        // B fragments (32x16 bf16): lane<16 -> K {0..7}u{16..23} of col; lane>=16 -> +8
        #pragma unroll
        for (int t = 0; t < 4; ++t) {
            Frag16 bfr;
            int col = (wn << 6) + (t << 4) + lh;
            int kb  = hi16 << 3;
            bfr.h[0] = *(const v8bf*)&Bs[buf][col][kb];
            bfr.h[1] = *(const v8bf*)&Bs[buf][col][kb + 16];
            acc[0][t] = __builtin_amdgcn_wmma_f32_16x16x32_bf16(false, af[0].v, false, bfr.v,
                                                                (short)0, acc[0][t], false, false);
            acc[1][t] = __builtin_amdgcn_wmma_f32_16x16x32_bf16(false, af[1].v, false, bfr.v,
                                                                (short)0, acc[1][t], false, false);
        }

        moe_wait_async();
        __syncthreads();
        buf ^= 1;
    }

    // combine: out[token] += 0.5*(acc + be[e]); each token is hit by exactly 2 experts
    const float* bep = be + (size_t)e * DIM;
    #pragma unroll
    for (int t = 0; t < 4; ++t) {
        int colb = n0 + (wn << 6) + (t << 4) + lh;
        float bb = bep[colb];
        #pragma unroll
        for (int s = 0; s < 2; ++s) {
            #pragma unroll
            for (int v = 0; v < 8; ++v) {
                int rloc = (wm << 5) + (s << 4) + (hi16 << 3) + v;
                if (rloc < valid) {
                    atomicAdd(&out[(size_t)toks[rloc] * DIM + colb],
                              0.5f * (acc[s][t][v] + bb));
                }
            }
        }
    }
}

// ---------------- launch ----------------
extern "C" void kernel_launch(void* const* d_in, const int* in_sizes, int n_in,
                              void* d_out, int out_size, void* d_ws, size_t ws_size,
                              hipStream_t stream) {
    (void)in_sizes; (void)n_in; (void)out_size; (void)ws_size;
    const float* x  = (const float*)d_in[0];
    const float* Wr = (const float*)d_in[1];
    const float* br = (const float*)d_in[2];
    const float* We = (const float*)d_in[3];
    const float* be = (const float*)d_in[4];
    float* out = (float*)d_out;

    char* ws = (char*)d_ws;
    __bf16* WeT   = (__bf16*)ws;                   // 20,971,520 B
    __bf16* xbf   = (__bf16*)(ws + 20971520);      // 16,777,216 B
    int* topk     = (int*)(ws + 37748736);         // 65,536 B
    int* counts   = (int*)(ws + 37814272);
    int* offs     = (int*)(ws + 37814336);
    int* cursors  = (int*)(ws + 37814400);
    int* tlist    = (int*)(ws + 37814464);

    moe_init_counts<<<1, 32, 0, stream>>>(counts);
    moe_we_transpose<<<NEXP * 256, 256, 0, stream>>>(We, WeT);
    moe_router<<<N_TOK / 8, 256, 0, stream>>>(x, Wr, br, xbf, topk, counts);
    moe_scan<<<1, 32, 0, stream>>>(counts, offs, cursors);
    moe_scatter<<<N_TOK / 256, 256, 0, stream>>>(topk, cursors, tlist);
    moe_zero_out<<<(N_TOK * DIM / 4) / 256, 256, 0, stream>>>((float4*)out);
    // worst case: one expert takes all 8192 tokens -> 64 M-tiles; 8 N-tiles; 10 experts
    moe_gemm<<<NEXP * 64 * 8, 256, 0, stream>>>(xbf, WeT, be, counts, offs, tlist, out);
}
